// Net_26938034880514
// MI455X (gfx1250) — matrix operand order repaired and checked by reading, not compiled, not run on hardware
//
#include <hip/hip_runtime.h>
#include <hip/hip_bf16.h>
#include <math.h>

#define N_NODES 50000
#define N_EDGES 400000
#define N_GRAPHS 512
#define TOWERS 5
#define F_IN 75
#define F_OUT 15
#define N_LAYERS 4
#define AVG_DEG_LOG 2.1972245773362196f
#define EPS_BN 1e-5f

// Packed K layout for the edge GEMM (all zero-padded in f16 packs):
//   [0,75)    h_dst   (pad 75..79)
//   [80,155)  h_src   (pad 155..159)
//   [160,210) e_raw   (pad 210..223)   => K_PACK = 224 = 7 tiles of 32
#define K_PACK 224
#define K_TILES 7
#define N_TILES 5          // 5 tiles of 16 cover F_IN=75 (padded to 80)
#define HP_W 80            // padded h row (halves)
#define EP_W 64            // padded e row (halves)
#define TOWER_HALVES (N_TILES * K_TILES * 512)   // 17920 halves = 35840 B per tower

typedef _Float16 half8  __attribute__((ext_vector_type(8)));
typedef _Float16 half16 __attribute__((ext_vector_type(16)));
typedef float    v8f    __attribute__((ext_vector_type(8)));

union H8 { uint4 u; _Float16 h[8]; };

// order-preserving float<->uint encoding for atomic min/max
__device__ __forceinline__ unsigned encf(float f) {
    unsigned u = __float_as_uint(f);
    return (u >> 31) ? ~u : (u | 0x80000000u);
}
__device__ __forceinline__ float decf(unsigned u) {
    u = (u >> 31) ? (u & 0x7fffffffu) : ~u;
    return __uint_as_float(u);
}
#define ENC_PINF 0xFF800000u   /* encf(+inf): min identity */
#define ENC_NINF 0x007FFFFFu   /* encf(-inf): max identity */

__device__ __forceinline__ void wait_asynccnt0() {
#if __has_builtin(__builtin_amdgcn_s_wait_asynccnt)
    __builtin_amdgcn_s_wait_asynccnt(0);
#else
    asm volatile("s_wait_asynccnt 0x0" ::: "memory");
#endif
}

// ---------------------------------------------------------------- pre-pass
__global__ void k_zero_int(int* p, int n) {
    int i = blockIdx.x * blockDim.x + threadIdx.x;
    if (i < n) p[i] = 0;
}
__global__ void k_deg(const int* __restrict__ ei, int* __restrict__ degI) {
    int e = blockIdx.x * blockDim.x + threadIdx.x;
    if (e < N_EDGES) atomicAdd(&degI[ei[N_EDGES + e]], 1);
}
__global__ void k_scalers(const int* __restrict__ degI, float* __restrict__ invdeg,
                          float* __restrict__ amp, float* __restrict__ att,
                          float* __restrict__ hasE) {
    int n = blockIdx.x * blockDim.x + threadIdx.x;
    if (n >= N_NODES) return;
    int raw = degI[n];
    float d = (float)(raw > 0 ? raw : 1);
    float ld = logf(d + 1.0f);
    invdeg[n] = 1.0f / d;
    amp[n]    = ld / AVG_DEG_LOG;
    att[n]    = AVG_DEG_LOG / ld;
    hasE[n]   = raw > 0 ? 1.0f : 0.0f;
}
// single-block inclusive scan over degree histogram -> CSR row starts
__global__ __launch_bounds__(1024)
void k_scan(const int* __restrict__ degI, int* __restrict__ rowStart, int* __restrict__ cursor) {
    __shared__ int s[1024];
    __shared__ int carry;
    if (threadIdx.x == 0) carry = 0;
    __syncthreads();
    for (int base = 0; base < N_NODES; base += 1024) {
        int i = base + (int)threadIdx.x;
        int v = (i < N_NODES) ? degI[i] : 0;
        s[threadIdx.x] = v;
        __syncthreads();
        for (int o = 1; o < 1024; o <<= 1) {
            int t = (threadIdx.x >= (unsigned)o) ? s[threadIdx.x - o] : 0;
            __syncthreads();
            s[threadIdx.x] += t;
            __syncthreads();
        }
        if (i < N_NODES) {
            rowStart[i] = carry + s[threadIdx.x] - v;   // exclusive
            cursor[i] = 0;
        }
        __syncthreads();
        if (threadIdx.x == 1023) carry += s[1023];
        __syncthreads();
    }
}
// counting-sort edges by destination node (graph is static across layers)
__global__ void k_sort(const int* __restrict__ ei, const int* __restrict__ rowStart,
                       int* __restrict__ cursor, int* __restrict__ srcS,
                       int* __restrict__ dstS, int* __restrict__ permE) {
    int e = blockIdx.x * blockDim.x + threadIdx.x;
    if (e >= N_EDGES) return;
    int d = ei[N_EDGES + e];
    int pos = rowStart[d] + atomicAdd(&cursor[d], 1);
    srcS[pos]  = ei[e];
    dstS[pos]  = d;
    permE[pos] = e;
}
__global__ void k_embed(const int* __restrict__ x, const float* __restrict__ node_emb,
                        float* __restrict__ h, _Float16* __restrict__ hpack) {
    int i = blockIdx.x * blockDim.x + threadIdx.x;
    if (i >= N_NODES * HP_W) return;
    int n = i / HP_W, c = i % HP_W;
    float v = (c < F_IN) ? node_emb[x[n] * F_IN + c] : 0.0f;
    hpack[i] = (_Float16)v;
    if (c < F_IN) h[n * F_IN + c] = v;
}
// edge embedding gathered in SORTED edge order
__global__ void k_epack(const int* __restrict__ eattr, const int* __restrict__ permE,
                        const float* __restrict__ edge_emb, _Float16* __restrict__ epackS) {
    int i = blockIdx.x * blockDim.x + threadIdx.x;
    if (i >= N_EDGES * EP_W) return;
    int e = i / EP_W, c = i % EP_W;
    float v = (c < 50) ? edge_emb[eattr[permE[e]] * 50 + c] : 0.0f;
    epackS[i] = (_Float16)v;
}

// ---------------------------------------------------------------- per-layer
__global__ void k_init_stats(float* __restrict__ sum, float* __restrict__ sumsq,
                             unsigned* __restrict__ mn, unsigned* __restrict__ mx,
                             float* __restrict__ bn_sum, float* __restrict__ bn_sumsq) {
    int i = blockIdx.x * blockDim.x + threadIdx.x;
    if (i >= N_NODES * TOWERS * F_IN) return;
    sum[i] = 0.0f; sumsq[i] = 0.0f;
    mn[i] = ENC_PINF; mx[i] = ENC_NINF;
    if (i < 128) { bn_sum[i] = 0.0f; bn_sumsq[i] = 0.0f; }
}

// Pre-swizzle folded weights P (K_PACK x 80 per tower) into per-lane v16h B fragments:
// element index = ((tile*32 + lane)*16 + j), tile = (t*5+nt)*7+kt, K=kt*32+(j<8 ? 8*(lane>>4)+j : 16+8*(lane>>4)+j-8), f=nt*16+(lane&15).
__global__ void k_pack_w(const float* __restrict__ pre_w,   // layer base (T,225,75)
                         const float* __restrict__ enc_w,   // layer base (50,75)
                         _Float16* __restrict__ Bp) {
    int idx = blockIdx.x * blockDim.x + threadIdx.x;
    const int total = TOWERS * N_TILES * K_TILES * 32 * 16;
    if (idx >= total) return;
    int j    = idx & 15;
    int lane = (idx >> 4) & 31;
    int tl   = idx >> 9;
    int kt   = tl % K_TILES;
    int tn   = tl / K_TILES;
    int nt   = tn % N_TILES;
    int t    = tn / N_TILES;
    int K = kt * 32 + ((j < 8) ? (8 * (lane >> 4) + j) : (16 + 8 * (lane >> 4) + (j - 8)));
    int f = nt * 16 + (lane & 15);
    float v = 0.0f;
    if (f < F_IN) {
        if (K < 75)        v = pre_w[(t * 225 + K) * F_IN + f];
        else if (K < 80)   v = 0.0f;
        else if (K < 155)  v = pre_w[(t * 225 + 75 + (K - 80)) * F_IN + f];
        else if (K < 160)  v = 0.0f;
        else if (K < 210) {
            int c = K - 160; float s = 0.0f;
            for (int jj = 0; jj < F_IN; ++jj)
                s += enc_w[c * F_IN + jj] * pre_w[(t * 225 + 150 + jj) * F_IN + f];
            v = s;
        }
    }
    Bp[idx] = (_Float16)v;
}
__global__ void k_fold_bias(const float* __restrict__ pre_w, const float* __restrict__ pre_b,
                            const float* __restrict__ enc_b, float* __restrict__ bfold) {
    int i = blockIdx.x * blockDim.x + threadIdx.x;
    if (i >= TOWERS * F_IN) return;
    int t = i / F_IN, f = i % F_IN;
    float s = pre_b[t * F_IN + f];
    for (int j = 0; j < F_IN; ++j)
        s += enc_b[j] * pre_w[(t * 225 + 150 + j) * F_IN + f];
    bfold[i] = s;
}

__device__ __forceinline__ uint4 ld_chunk(const _Float16* rD, const _Float16* rS,
                                          const _Float16* rE, int kk) {
    const _Float16* p;
    if (kk < 80)       p = rD + kk;
    else if (kk < 160) p = rS + (kk - 80);
    else               p = rE + (kk - 160);
    return *(const uint4*)p;
}
__device__ __forceinline__ void flush_stats(int node, int t, int f, float aS, float aQ,
                                            float aMn, float aMx,
                                            float* __restrict__ sum, float* __restrict__ sumsq,
                                            unsigned* __restrict__ mn, unsigned* __restrict__ mx) {
    size_t o = (size_t)node * (TOWERS * F_IN) + t * F_IN + f;
    atomicAdd(&sum[o], aS);
    atomicAdd(&sumsq[o], aQ);
    atomicMin(&mn[o], encf(aMn));
    atomicMax(&mx[o], encf(aMx));
}

// 8 waves per block; each wave owns a 16-edge (dst-sorted) tile.
// Per tower: async DMA of that tower's swizzled B panel (35 KB) straight into LDS
// (GLOBAL_LOAD_ASYNC_TO_LDS_B128, ASYNCcnt), then 5 N-tiles x 7 WMMA per wave with
// register-preloaded B fragments and dual accumulators, followed by run-length-
// aggregated atomic scatter.
__global__ __launch_bounds__(256)
void k_edge_msg(const int* __restrict__ srcS, const int* __restrict__ dstS,
                const _Float16* __restrict__ hpack, const _Float16* __restrict__ epackS,
                const _Float16* __restrict__ Bp, const float* __restrict__ bfold,
                float* __restrict__ sum, float* __restrict__ sumsq,
                unsigned* __restrict__ mn, unsigned* __restrict__ mx) {
    __shared__ __align__(32) _Float16 sB[TOWER_HALVES];
    const int lane = threadIdx.x & 31;
    const int wave = threadIdx.x >> 5;
    const int e0   = (blockIdx.x * 8 + wave) * 16;
    const int mrow = lane & 15;
    const int off  = (lane >= 16) ? 8 : 0;
    const _Float16* rowD = hpack  + (size_t)dstS[e0 + mrow] * HP_W;
    const _Float16* rowS = hpack  + (size_t)srcS[e0 + mrow] * HP_W;
    const _Float16* rowE = epackS + (size_t)(e0 + mrow) * EP_W;

    // A fragments (ISA 16-bit 16x32 layout): per lane, halves 0..7 = K off..off+7,
    // halves 8..15 = K off+16..off+23 -> two aligned 16B loads per K-tile.
    half16 A[K_TILES];
#pragma unroll
    for (int kt = 0; kt < K_TILES; ++kt) {
        int k0 = kt * 32 + off;
        H8 c0, c1;
        c0.u = ld_chunk(rowD, rowS, rowE, k0);
        c1.u = ld_chunk(rowD, rowS, rowE, k0 + 16);
#pragma unroll
        for (int i = 0; i < 8; ++i) { A[kt][i] = c0.h[i]; A[kt][8 + i] = c1.h[i]; }
    }

    const int fl = lane & 15;
#pragma unroll 1
    for (int t = 0; t < TOWERS; ++t) {
        __syncthreads();                       // previous tower's sB fully consumed
        {
            // async DMA: memory -> LDS, 16B per lane per issue, tracked by ASYNCcnt
            const _Float16* gsrc = Bp + (size_t)t * TOWER_HALVES;
            for (int i = threadIdx.x; i < TOWER_HALVES / 8; i += 256) {
                unsigned lds_addr = (unsigned)(size_t)(sB + (size_t)i * 8);
                unsigned long long gaddr = (unsigned long long)(size_t)(gsrc + (size_t)i * 8);
                asm volatile("global_load_async_to_lds_b128 %0, %1, off"
                             :: "v"(lds_addr), "v"(gaddr) : "memory");
            }
            wait_asynccnt0();
        }
        __syncthreads();
#pragma unroll 1
        for (int nt = 0; nt < N_TILES; ++nt) {
            // preload all 7 B fragments, then run the WMMA chain (dual accumulators)
            half16 Bf[K_TILES];
#pragma unroll
            for (int kt = 0; kt < K_TILES; ++kt)
                Bf[kt] = *(const half16*)(sB + (size_t)((nt * K_TILES + kt) * 32 + lane) * 16);
            v8f acc0 = {0.f, 0.f, 0.f, 0.f, 0.f, 0.f, 0.f, 0.f};
            v8f acc1 = {0.f, 0.f, 0.f, 0.f, 0.f, 0.f, 0.f, 0.f};
#pragma unroll
            for (int kt = 0; kt < K_TILES; kt += 2)
                acc0 = __builtin_amdgcn_wmma_f32_16x16x32_f16(
                    false, A[kt], false, Bf[kt], (short)0, acc0, false, false);
#pragma unroll
            for (int kt = 1; kt < K_TILES; kt += 2)
                acc1 = __builtin_amdgcn_wmma_f32_16x16x32_f16(
                    false, A[kt], false, Bf[kt], (short)0, acc1, false, false);
            v8f acc = acc0 + acc1;

            int f = nt * 16 + fl;
            if (f < F_IN) {
                float bf = bfold[t * F_IN + f];
                // dst-sorted rows: aggregate runs in registers, flush on node change
                int   cur = dstS[e0 + off];
                float aS = 0.f, aQ = 0.f, aMn = 3.402823e38f, aMx = -3.402823e38f;
#pragma unroll
                for (int v = 0; v < 8; ++v) {
                    int node = dstS[e0 + off + v];
                    if (node != cur) {
                        flush_stats(cur, t, f, aS, aQ, aMn, aMx, sum, sumsq, mn, mx);
                        cur = node; aS = 0.f; aQ = 0.f; aMn = 3.402823e38f; aMx = -3.402823e38f;
                    }
                    float val = acc[v] + bf;
                    aS += val; aQ += val * val;
                    aMn = fminf(aMn, val); aMx = fmaxf(aMx, val);
                }
                flush_stats(cur, t, f, aS, aQ, aMn, aMx, sum, sumsq, mn, mx);
            }
        }
    }
}

// post-MLP: out[n,t,fo] = xt@Wx + sum_s agg_s @ (W1 + amp*W2 + att*W3) + b
__global__ void k_node_post(const float* __restrict__ h,
                            const float* __restrict__ sum, const float* __restrict__ sumsq,
                            const unsigned* __restrict__ mn, const unsigned* __restrict__ mx,
                            const float* __restrict__ invdeg, const float* __restrict__ hasE,
                            const float* __restrict__ amp, const float* __restrict__ att,
                            const float* __restrict__ post_w,  // layer base (T,975,15)
                            const float* __restrict__ post_b,  // layer base (T,15)
                            float* __restrict__ ht) {
    int idx = blockIdx.x * blockDim.x + threadIdx.x;
    if (idx >= N_NODES * F_IN) return;
    int n = idx / F_IN, r = idx % F_IN;
    int t = r / F_OUT, fo = r % F_OUT;
    float amp_ = amp[n], att_ = att[n], inv = invdeg[n], he = hasE[n];
    const float* W = post_w + (size_t)t * 975 * F_OUT;
    const float* hn = h + (size_t)n * F_IN;
    float acc = post_b[t * F_OUT + fo];
#pragma unroll 5
    for (int c = 0; c < F_IN; ++c) acc += hn[c] * W[c * F_OUT + fo];
    size_t ab = (size_t)n * (TOWERS * F_IN) + t * F_IN;
    for (int c = 0; c < F_IN; ++c) {
        float mean = sum[ab + c] * inv;
        float sq   = sumsq[ab + c] * inv;
        float sd   = sqrtf(fmaxf(sq - mean * mean, 0.0f) + 1e-5f);
        float vmn  = (he > 0.0f) ? decf(mn[ab + c]) : 0.0f;
        float vmx  = (he > 0.0f) ? decf(mx[ab + c]) : 0.0f;
        float vals[4] = {mean, vmn, vmx, sd};
#pragma unroll
        for (int s = 0; s < 4; ++s) {
            int r1 = F_IN + s * F_IN + c;
            float w = W[r1 * F_OUT + fo] + amp_ * W[(r1 + 300) * F_OUT + fo]
                                         + att_ * W[(r1 + 600) * F_OUT + fo];
            acc += vals[s] * w;
        }
    }
    ht[(size_t)n * F_IN + t * F_OUT + fo] = acc;
}

__global__ void k_lin_bn(const float* __restrict__ ht, const float* __restrict__ lin_w,
                         const float* __restrict__ lin_b, float* __restrict__ hc,
                         float* __restrict__ bn_sum, float* __restrict__ bn_sumsq) {
    int idx = blockIdx.x * blockDim.x + threadIdx.x;
    if (idx >= N_NODES * F_IN) return;
    int n = idx / F_IN, f = idx % F_IN;
    const float* hr = ht + (size_t)n * F_IN;
    float acc = lin_b[f];
#pragma unroll 5
    for (int c = 0; c < F_IN; ++c) acc += hr[c] * lin_w[c * F_IN + f];
    hc[idx] = acc;
    atomicAdd(&bn_sum[f], acc);
    atomicAdd(&bn_sumsq[f], acc * acc);
}
__global__ void k_bn_stats(const float* __restrict__ bn_sum, const float* __restrict__ bn_sumsq,
                           const float* __restrict__ gamma, const float* __restrict__ beta,
                           float* __restrict__ bn_scale, float* __restrict__ bn_shift) {
    int f = threadIdx.x;
    if (f >= F_IN) return;
    float mu  = bn_sum[f] * (1.0f / N_NODES);
    float var = bn_sumsq[f] * (1.0f / N_NODES) - mu * mu;
    float sc  = gamma[f] * rsqrtf(var + EPS_BN);
    bn_scale[f] = sc;
    bn_shift[f] = beta[f] - mu * sc;
}
__global__ void k_bn_apply(const float* __restrict__ hc, const float* __restrict__ bn_scale,
                           const float* __restrict__ bn_shift, float* __restrict__ h,
                           _Float16* __restrict__ hpack) {
    int idx = blockIdx.x * blockDim.x + threadIdx.x;
    if (idx >= N_NODES * F_IN) return;
    int n = idx / F_IN, f = idx % F_IN;
    float v = fmaxf(hc[idx] * bn_scale[f] + bn_shift[f], 0.0f);
    h[idx] = v;
    hpack[(size_t)n * HP_W + f] = (_Float16)v;
}

// ---------------------------------------------------------------- readout
__global__ void k_zero_f(float* p, int n) {
    int i = blockIdx.x * blockDim.x + threadIdx.x;
    if (i < n) p[i] = 0.0f;
}
__global__ void k_pool(const float* __restrict__ h, const int* __restrict__ batch,
                       float* __restrict__ g) {
    int idx = blockIdx.x * blockDim.x + threadIdx.x;
    if (idx >= N_NODES * F_IN) return;
    int n = idx / F_IN, f = idx % F_IN;
    atomicAdd(&g[batch[n] * F_IN + f], h[idx]);
}
__global__ void k_mlp(const float* __restrict__ g,
                      const float* __restrict__ w1, const float* __restrict__ b1,
                      const float* __restrict__ w2, const float* __restrict__ b2,
                      const float* __restrict__ w3, const float* __restrict__ b3,
                      float* __restrict__ out) {
    int gi = blockIdx.x * blockDim.x + threadIdx.x;
    if (gi >= N_GRAPHS) return;
    const float* gr = g + (size_t)gi * F_IN;
    float a1[50];
#pragma unroll
    for (int o = 0; o < 50; ++o) a1[o] = b1[o];
    for (int c = 0; c < F_IN; ++c) {
        float x = gr[c];
#pragma unroll
        for (int o = 0; o < 50; ++o) a1[o] += x * w1[c * 50 + o];
    }
    float a2[25];
#pragma unroll
    for (int o = 0; o < 25; ++o) a2[o] = b2[o];
#pragma unroll
    for (int c = 0; c < 50; ++c) {
        float x = fmaxf(a1[c], 0.0f);
#pragma unroll
        for (int o = 0; o < 25; ++o) a2[o] += x * w2[c * 25 + o];
    }
    float r = b3[0];
#pragma unroll
    for (int c = 0; c < 25; ++c) r += fmaxf(a2[c], 0.0f) * w3[c];
    out[gi] = r;
}

// ---------------------------------------------------------------- launch
extern "C" void kernel_launch(void* const* d_in, const int* in_sizes, int n_in,
                              void* d_out, int out_size, void* d_ws, size_t ws_size,
                              hipStream_t stream) {
    const int*   x        = (const int*)d_in[0];
    const int*   ei       = (const int*)d_in[1];
    const int*   eattr    = (const int*)d_in[2];
    const int*   batch    = (const int*)d_in[3];
    const float* node_emb = (const float*)d_in[4];
    const float* edge_emb = (const float*)d_in[5];
    const float* enc_w    = (const float*)d_in[6];
    const float* enc_b    = (const float*)d_in[7];
    const float* pre_w    = (const float*)d_in[8];
    const float* pre_b    = (const float*)d_in[9];
    const float* post_w   = (const float*)d_in[10];
    const float* post_b   = (const float*)d_in[11];
    const float* lin_w    = (const float*)d_in[12];
    const float* lin_b    = (const float*)d_in[13];
    const float* bn_g     = (const float*)d_in[14];
    const float* bn_b     = (const float*)d_in[15];
    const float* m1w      = (const float*)d_in[16];
    const float* m1b      = (const float*)d_in[17];
    const float* m2w      = (const float*)d_in[18];
    const float* m2b      = (const float*)d_in[19];
    const float* m3w      = (const float*)d_in[20];
    const float* m3b      = (const float*)d_in[21];
    float* out = (float*)d_out;

    char* p = (char*)d_ws;
    auto alloc = [&](size_t bytes) { char* r = p; p += ((bytes + 255) / 256) * 256; return r; };
    const size_t NS = (size_t)N_NODES * TOWERS * F_IN;
    float*     h       = (float*)    alloc((size_t)N_NODES * F_IN * 4);
    _Float16*  hpack   = (_Float16*) alloc((size_t)N_NODES * HP_W * 2);
    _Float16*  epackS  = (_Float16*) alloc((size_t)N_EDGES * EP_W * 2);
    float*     sum     = (float*)    alloc(NS * 4);
    float*     sumsq   = (float*)    alloc(NS * 4);
    unsigned*  mn      = (unsigned*) alloc(NS * 4);
    unsigned*  mx      = (unsigned*) alloc(NS * 4);
    float*     ht      = (float*)    alloc((size_t)N_NODES * F_IN * 4);
    float*     hc      = (float*)    alloc((size_t)N_NODES * F_IN * 4);
    _Float16*  Bpack   = (_Float16*) alloc((size_t)TOWERS * TOWER_HALVES * 2);
    float*     bfold   = (float*)    alloc(TOWERS * F_IN * 4);
    int*       degI    = (int*)      alloc((size_t)N_NODES * 4);
    int*       rowStart= (int*)      alloc((size_t)(N_NODES + 1) * 4);
    int*       cursor  = (int*)      alloc((size_t)N_NODES * 4);
    int*       srcS    = (int*)      alloc((size_t)N_EDGES * 4);
    int*       dstS    = (int*)      alloc((size_t)N_EDGES * 4);
    int*       permE   = (int*)      alloc((size_t)N_EDGES * 4);
    float*     invdeg  = (float*)    alloc((size_t)N_NODES * 4);
    float*     amp     = (float*)    alloc((size_t)N_NODES * 4);
    float*     att     = (float*)    alloc((size_t)N_NODES * 4);
    float*     hasE    = (float*)    alloc((size_t)N_NODES * 4);
    float*     bn_sum  = (float*)    alloc(128 * 4);
    float*     bn_sumsq= (float*)    alloc(128 * 4);
    float*     bn_scale= (float*)    alloc(128 * 4);
    float*     bn_shift= (float*)    alloc(128 * 4);
    float*     g       = (float*)    alloc((size_t)N_GRAPHS * F_IN * 4);

    auto nb = [](long n, int b) { return (int)((n + b - 1) / b); };

    // ---- pre-pass: degrees, scalers, CSR sort of edges by dst, packed embeddings
    k_zero_int<<<nb(N_NODES, 256), 256, 0, stream>>>(degI, N_NODES);
    k_deg<<<nb(N_EDGES, 256), 256, 0, stream>>>(ei, degI);
    k_scalers<<<nb(N_NODES, 256), 256, 0, stream>>>(degI, invdeg, amp, att, hasE);
    k_scan<<<1, 1024, 0, stream>>>(degI, rowStart, cursor);
    k_sort<<<nb(N_EDGES, 256), 256, 0, stream>>>(ei, rowStart, cursor, srcS, dstS, permE);
    k_embed<<<nb((long)N_NODES * HP_W, 256), 256, 0, stream>>>(x, node_emb, h, hpack);
    k_epack<<<nb((long)N_EDGES * EP_W, 256), 256, 0, stream>>>(eattr, permE, edge_emb, epackS);

    // ---- layers
    for (int i = 0; i < N_LAYERS; ++i) {
        const float* l_enc_w  = enc_w  + (size_t)i * 50 * F_IN;
        const float* l_enc_b  = enc_b  + (size_t)i * F_IN;
        const float* l_pre_w  = pre_w  + (size_t)i * TOWERS * 225 * F_IN;
        const float* l_pre_b  = pre_b  + (size_t)i * TOWERS * F_IN;
        const float* l_post_w = post_w + (size_t)i * TOWERS * 975 * F_OUT;
        const float* l_post_b = post_b + (size_t)i * TOWERS * F_OUT;
        const float* l_lin_w  = lin_w  + (size_t)i * F_IN * F_IN;
        const float* l_lin_b  = lin_b  + (size_t)i * F_IN;
        const float* l_bn_g   = bn_g   + (size_t)i * F_IN;
        const float* l_bn_b   = bn_b   + (size_t)i * F_IN;

        k_init_stats<<<nb((long)NS, 256), 256, 0, stream>>>(sum, sumsq, mn, mx, bn_sum, bn_sumsq);
        k_pack_w<<<nb(TOWERS * N_TILES * K_TILES * 512, 256), 256, 0, stream>>>(l_pre_w, l_enc_w, Bpack);
        k_fold_bias<<<nb(TOWERS * F_IN, 128), 128, 0, stream>>>(l_pre_w, l_pre_b, l_enc_b, bfold);
        k_edge_msg<<<(N_EDGES / 16) / 8, 256, 0, stream>>>(srcS, dstS, hpack, epackS, Bpack,
                                                           bfold, sum, sumsq, mn, mx);
        k_node_post<<<nb((long)N_NODES * F_IN, 256), 256, 0, stream>>>(
            h, sum, sumsq, mn, mx, invdeg, hasE, amp, att, l_post_w, l_post_b, ht);
        k_lin_bn<<<nb((long)N_NODES * F_IN, 256), 256, 0, stream>>>(ht, l_lin_w, l_lin_b, hc, bn_sum, bn_sumsq);
        k_bn_stats<<<1, 128, 0, stream>>>(bn_sum, bn_sumsq, l_bn_g, l_bn_b, bn_scale, bn_shift);
        k_bn_apply<<<nb((long)N_NODES * F_IN, 256), 256, 0, stream>>>(hc, bn_scale, bn_shift, h, hpack);
    }

    // ---- readout
    k_zero_f<<<nb(N_GRAPHS * F_IN, 256), 256, 0, stream>>>(g, N_GRAPHS * F_IN);
    k_pool<<<nb((long)N_NODES * F_IN, 256), 256, 0, stream>>>(h, batch, g);
    k_mlp<<<nb(N_GRAPHS, 256), 256, 0, stream>>>(g, m1w, m1b, m2w, m2b, m3w, m3b, out);
}
